// ScaledDotProductAttention_70884140253872
// MI455X (gfx1250) — matrix operand order repaired and checked by reading, not compiled, not run on hardware
//
#include <hip/hip_runtime.h>

typedef __attribute__((ext_vector_type(16))) _Float16 v16h;
typedef __attribute__((ext_vector_type(8)))  float    v8f;
typedef __attribute__((ext_vector_type(4)))  float    v4f;

#define SEQ   2048
#define DIM   64
#define QTILE 16

// One workgroup (8 wave32s) per (batch*head, 16-query tile).
// LDS: full 16 x 2048 fp32 score strip (128KB; 2 workgroups fit in CDNA5's
// 320KB WGP LDS) so exp-scores are computed once, row-summed, normalized and
// streamed to HBM exactly once with non-temporal stores.
__global__ __launch_bounds__(256) void attn_fwd_kernel(
    const float* __restrict__ Q,
    const float* __restrict__ K,
    const float* __restrict__ V,
    float* __restrict__ ctx_out,
    float* __restrict__ attn_out)
{
    extern __shared__ char smem_raw[];
    float* scoresf = (float*)smem_raw;            // [16][2048] unnormalized exp-scores
    float* psum    = scoresf + QTILE * SEQ;       // [16][16] partial row sums
    float* invr    = psum + 256;                  // [16] 1/(rowsum+eps)
    float* ctxs    = invr + 16;                   // [16][64] context accumulator

    const int t    = threadIdx.x;
    const int wave = t >> 5;
    const int lane = t & 31;
    const int g    = lane >> 4;   // lane half-group (WMMA K-range select)
    const int rn   = lane & 15;   // A-row / B-column index

    const int gid = blockIdx.x;
    const int bh  = gid >> 7;     // / 128 query tiles
    const int q0  = (gid & 127) * QTILE;

    const float* Qb = Q + ((size_t)bh * SEQ + q0) * DIM;
    const float* Kb = K + (size_t)bh * SEQ * DIM;
    const float* Vb = V + (size_t)bh * SEQ * DIM;
    float* attn_strip = attn_out + (size_t)bh * SEQ * SEQ + (size_t)q0 * SEQ;

    // ---------------- Phase 1: scores = exp(Q K^T / 8) into LDS ----------------
    // A fragments (16x32 f16): lanes 0-15 hold K = e (e<8) / 16+(e-8) ; lanes 16-31 +8
    const float* qrow = Qb + (size_t)rn * DIM;
    v16h a0, a1;
#pragma unroll
    for (int e = 0; e < 8; ++e) {
        a0[e]     = (_Float16)qrow[      g * 8 + e];
        a0[8 + e] = (_Float16)qrow[16 +  g * 8 + e];
        a1[e]     = (_Float16)qrow[32 +       g * 8 + e];
        a1[8 + e] = (_Float16)qrow[32 + 16 +  g * 8 + e];
    }

    for (int i = 0; i < 16; ++i) {
        const int key0 = (wave * 16 + i) * 16;
        const float* krow = Kb + (size_t)(key0 + rn) * DIM;
        if (i + 1 < 16) __builtin_prefetch(krow + 16 * DIM, 0, 0);  // next key tile

        // B fragments (32x16 f16): lane group g holds K = g*8 + e (e<8) / 16+g*8+(e-8), column = rn
        v16h b0, b1;
#pragma unroll
        for (int e = 0; e < 8; ++e) {
            b0[e]     = (_Float16)krow[      g * 8 + e];
            b0[8 + e] = (_Float16)krow[16 +  g * 8 + e];
            b1[e]     = (_Float16)krow[32 +  g * 8 + e];
            b1[8 + e] = (_Float16)krow[48 +  g * 8 + e];
        }
        v8f acc = {};
        acc = __builtin_amdgcn_wmma_f32_16x16x32_f16(false, a0, false, b0,
                                                     (short)0, acc, false, false);
        acc = __builtin_amdgcn_wmma_f32_16x16x32_f16(false, a1, false, b1,
                                                     (short)0, acc, false, false);
#pragma unroll
        for (int v = 0; v < 8; ++v) {
            const int m = g * 8 + v;                       // D layout: row = v (+8 for hi lanes)
            scoresf[m * SEQ + key0 + rn] = __expf(acc[v] * 0.125f);
        }
    }
    __syncthreads();

    // ---------------- Phase 2: row sums (v4f LDS reads) + zero context accum ----------
    {
        const int row = t >> 4, seg = t & 15;
        const v4f* p = (const v4f*)(scoresf + row * SEQ + seg * 128);
        v4f s4 = {};
#pragma unroll 8
        for (int k2 = 0; k2 < 32; ++k2) s4 += p[k2];
        psum[row * 16 + seg] = s4.x + s4.y + s4.z + s4.w;
        for (int l = t; l < QTILE * DIM; l += 256) ctxs[l] = 0.f;
    }
    __syncthreads();
    if (t < 16) {
        float s = 0.f;
#pragma unroll
        for (int k2 = 0; k2 < 16; ++k2) s += psum[t * 16 + k2];
        invr[t] = 1.f / (s + 1e-8f);
    }
    __syncthreads();

    // ------- attn output: coalesced 16B non-temporal streaming stores -------
    // (write-once data: NT keeps 805MB of dead writes from evicting K/V in L2)
    {
        const v4f* src = (const v4f*)scoresf;
        v4f* dst = (v4f*)attn_strip;
        for (int l4 = t; l4 < QTILE * SEQ / 4; l4 += 256) {
            const int row = l4 >> 9;               // 512 v4f per row
            v4f v = src[l4] * invr[row];
            __builtin_nontemporal_store(v, dst + l4);
        }
    }

    // ---------------- Phase 3: context = attn * V ----------------
    {
        const int n0 = (wave & 3) * 16;            // output dim tile
        const float inv_r = invr[rn];              // scale for A rows
        v8f acc = {};
        for (int c = (wave >> 2); c < 64; c += 2) {   // 64 K-chunks of 32 keys
            const int cb = c * 32;
            if (c + 2 < 64)
                __builtin_prefetch(Vb + (size_t)(cb + 64 + g * 16) * DIM + n0 + rn, 0, 0);
            v16h a, b;
#pragma unroll
            for (int e = 0; e < 8; ++e) {
                a[e]     = (_Float16)(scoresf[rn * SEQ + cb +      g * 8 + e] * inv_r);
                a[8 + e] = (_Float16)(scoresf[rn * SEQ + cb + 16 + g * 8 + e] * inv_r);
            }
#pragma unroll
            for (int e = 0; e < 16; ++e)
                b[e] = (_Float16)Vb[(size_t)(cb + (e < 8 ? g * 8 + e : 16 + g * 8 + (e - 8))) * DIM + n0 + rn];
            acc = __builtin_amdgcn_wmma_f32_16x16x32_f16(false, a, false, b,
                                                         (short)0, acc, false, false);
        }
#pragma unroll
        for (int v = 0; v < 8; ++v)
            atomicAdd(&ctxs[(g * 8 + v) * DIM + n0 + rn], acc[v]);
    }
    __syncthreads();

    // ------- context output: one 16B NT store per thread -------
    {
        const int row = t >> 4, d4 = t & 15;       // 256 threads == 16 rows x 16 v4f
        v4f v = ((const v4f*)ctxs)[t];
        v4f* dst = (v4f*)(ctx_out + ((size_t)bh * SEQ + q0 + row) * DIM) + d4;
        __builtin_nontemporal_store(v, dst);
    }
}

extern "C" void kernel_launch(void* const* d_in, const int* in_sizes, int n_in,
                              void* d_out, int out_size, void* d_ws, size_t ws_size,
                              hipStream_t stream) {
    (void)in_sizes; (void)n_in; (void)d_ws; (void)ws_size; (void)out_size;
    const float* Q = (const float*)d_in[0];
    const float* K = (const float*)d_in[1];
    const float* V = (const float*)d_in[2];

    constexpr int B = 4, H = 12;
    float* ctx_out  = (float*)d_out;                               // [B,H,S,D] first
    float* attn_out = ctx_out + (size_t)B * H * SEQ * DIM;         // then [B,H,S,S]

    const size_t shmem =
        (size_t)(QTILE * SEQ + 256 + 16 + QTILE * DIM) * sizeof(float); // ~136 KB
    static_assert(QTILE * SEQ * sizeof(float) == 131072, "score strip size");

    (void)hipFuncSetAttribute((const void*)attn_fwd_kernel,
                              hipFuncAttributeMaxDynamicSharedMemorySize,
                              (int)shmem);

    dim3 grid(B * H * (SEQ / QTILE));   // 6144 workgroups
    attn_fwd_kernel<<<grid, 256, shmem, stream>>>(Q, K, V, ctx_out, attn_out);
}
